// myDecoder_expandedVersion_80796924773034
// MI455X (gfx1250) — matrix operand order repaired and checked by reading, not compiled, not run on hardware
//
#include <hip/hip_runtime.h>
#include <hip/hip_bf16.h>
#include <math.h>

// Problem dimensions (fixed by the reference)
#define Bq  64
#define Sq  128
#define Tq  48
#define Hq  512
#define EHq 1024
#define Vq  32000

typedef __bf16 bf16_t;
typedef __attribute__((ext_vector_type(16))) __bf16 v16bf;
typedef __attribute__((ext_vector_type(8)))  float  v8f;

__device__ __forceinline__ bf16_t f2bf(float f) {
  union { float f; unsigned u; } x; x.f = f;
  unsigned r = x.u + 0x7FFFu + ((x.u >> 16) & 1u);   // round-to-nearest-even
  unsigned short hs = (unsigned short)(r >> 16);
  union { unsigned short s; bf16_t b; } y; y.s = hs;
  return y.b;
}

__device__ __forceinline__ float sigf(float x) { return 1.0f / (1.0f + expf(-x)); }

union Frag { v16bf v; uint4 q[2]; };

// ---------------------------------------------------------------------------
// Big-GEMM path: LDS-staged, double-buffered bf16 WMMA GEMM.
//   C(MxN, f32) = act( A(MxK, bf16) * W(NxK, bf16)^T + bias )
// block = 256 threads (8 waves, 4(M) x 2(N)); block tile 128x128;
// wave tile 32x64 -> 8 wmmas per K-step. K staged in 32-wide slabs in LDS,
// rows padded to 40 bf16 (80B) for conflict-free ds_load_b128 fragments.
// Requires M%128==0, N%128==0, K%32==0.
// ---------------------------------------------------------------------------
__global__ __launch_bounds__(256) void wmma_gemm_lds_kernel(
    const bf16_t* __restrict__ A, const bf16_t* __restrict__ W,
    const float* __restrict__ bias, float* __restrict__ C,
    int M, int N, int K, int act)
{
  __shared__ bf16_t As[2][128][40];
  __shared__ bf16_t Bs[2][128][40];

  const int tid  = threadIdx.x;
  const int lane = tid & 31;
  const int wv   = tid >> 5;        // 0..7
  const int wm   = wv >> 1;         // 0..3  (M direction)
  const int wn   = wv & 1;          // 0..1  (N direction)
  const int r    = lane & 15;
  const int hf   = lane >> 4;
  const int rowBlock = blockIdx.y * 128;
  const int colBlock = blockIdx.x * 128;

  // cooperative global->LDS staging: 256 threads x 32B each per tile
  const int ldRow = tid >> 1;               // 0..127
  const int ldSeg = (tid & 1) * 16;         // bf16 offset within 32-wide slab

  const bf16_t* aSrc = A + (size_t)(rowBlock + ldRow) * K + ldSeg;
  const bf16_t* wSrc = W + (size_t)(colBlock + ldRow) * K + ldSeg;

  v8f acc[2][4] = {};

  uint4 ra0, ra1, rb0, rb1;
  // prologue: stage k-tile 0
  ra0 = *(const uint4*)(aSrc);
  ra1 = *(const uint4*)(aSrc + 8);
  rb0 = *(const uint4*)(wSrc);
  rb1 = *(const uint4*)(wSrc + 8);
  *(uint4*)&As[0][ldRow][ldSeg]     = ra0;
  *(uint4*)&As[0][ldRow][ldSeg + 8] = ra1;
  *(uint4*)&Bs[0][ldRow][ldSeg]     = rb0;
  *(uint4*)&Bs[0][ldRow][ldSeg + 8] = rb1;
  __syncthreads();

  const int nk = K >> 5;
  for (int kt = 0; kt < nk; ++kt) {
    const int buf = kt & 1;
    const bool more = (kt + 1) < nk;
    if (more) {
      // issue next slab's global loads BEFORE compute; wait lands on the
      // LDS stores after the wmmas -> latency overlapped with matrix math
      const bf16_t* an = aSrc + (size_t)(kt + 1) * 32;
      const bf16_t* bn = wSrc + (size_t)(kt + 1) * 32;
      ra0 = *(const uint4*)(an);
      ra1 = *(const uint4*)(an + 8);
      rb0 = *(const uint4*)(bn);
      rb1 = *(const uint4*)(bn + 8);
    }

    Frag af[2];
#pragma unroll
    for (int mt = 0; mt < 2; ++mt) {
      const int arow = wm * 32 + mt * 16 + r;
      af[mt].q[0] = *(const uint4*)&As[buf][arow][hf * 8];
      af[mt].q[1] = *(const uint4*)&As[buf][arow][16 + hf * 8];
    }
#pragma unroll
    for (int nt = 0; nt < 4; ++nt) {
      const int brow = wn * 64 + nt * 16 + r;
      Frag bfr;
      bfr.q[0] = *(const uint4*)&Bs[buf][brow][hf * 8];
      bfr.q[1] = *(const uint4*)&Bs[buf][brow][16 + hf * 8];
      acc[0][nt] = __builtin_amdgcn_wmma_f32_16x16x32_bf16(
          false, af[0].v, false, bfr.v, (short)0, acc[0][nt], false, false);
      acc[1][nt] = __builtin_amdgcn_wmma_f32_16x16x32_bf16(
          false, af[1].v, false, bfr.v, (short)0, acc[1][nt], false, false);
    }

    if (more) {
      const int nb = buf ^ 1;
      *(uint4*)&As[nb][ldRow][ldSeg]     = ra0;
      *(uint4*)&As[nb][ldRow][ldSeg + 8] = ra1;
      *(uint4*)&Bs[nb][ldRow][ldSeg]     = rb0;
      *(uint4*)&Bs[nb][ldRow][ldSeg + 8] = rb1;
    }
    __syncthreads();
  }

  // C/D layout: elem j -> M = tileRow + hf*8 + j, N = tileCol + (lane&15)
#pragma unroll
  for (int mt = 0; mt < 2; ++mt) {
    const int mB = rowBlock + wm * 32 + mt * 16 + hf * 8;
#pragma unroll
    for (int nt = 0; nt < 4; ++nt) {
      const int n = colBlock + wn * 64 + nt * 16 + r;
      const float bs = bias ? bias[n] : 0.0f;
      float* cp = C + (size_t)mB * N + n;
#pragma unroll
      for (int j = 0; j < 8; ++j) {
        float v = acc[mt][nt][j] + bs;
        if (act) v = tanhf(v);
        cp[(size_t)j * N] = v;
      }
    }
  }
}

// ---------------------------------------------------------------------------
// Small-GEMM path (M=64 recurrent steps): register-resident WMMA GEMM.
// block = 128 threads (4 waves along M); block tile 64x64; wave tile 16x64.
// ---------------------------------------------------------------------------
__global__ __launch_bounds__(128) void wmma_gemm_kernel(
    const bf16_t* __restrict__ A, const bf16_t* __restrict__ W,
    const float* __restrict__ bias, float* __restrict__ C,
    int M, int N, int K, int act)
{
  const int lane = threadIdx.x & 31;
  const int wv   = threadIdx.x >> 5;
  const int r    = lane & 15;
  const int hf   = lane >> 4;
  const int rowBase = blockIdx.y * 64 + wv * 16;
  const int colBase = blockIdx.x * 64;

  v8f acc[4] = {};

  const bf16_t* pa = A + (size_t)(rowBase + r) * K;
  for (int k0 = 0; k0 < K; k0 += 32) {
    Frag af;
    af.q[0] = *(const uint4*)(pa + k0 + hf * 8);
    af.q[1] = *(const uint4*)(pa + k0 + 16 + hf * 8);
    if (k0 + 32 < K) __builtin_prefetch(pa + k0 + 32, 0, 0);
#pragma unroll
    for (int nt = 0; nt < 4; ++nt) {
      const bf16_t* pw = W + (size_t)(colBase + nt * 16 + r) * K + k0 + hf * 8;
      Frag wf;
      wf.q[0] = *(const uint4*)(pw);
      wf.q[1] = *(const uint4*)(pw + 16);
      acc[nt] = __builtin_amdgcn_wmma_f32_16x16x32_bf16(
          false, af.v, false, wf.v, (short)0, acc[nt], false, false);
    }
  }

  const int mB = rowBase + hf * 8;
#pragma unroll
  for (int nt = 0; nt < 4; ++nt) {
    const int n = colBase + nt * 16 + r;
    const float bs = bias ? bias[n] : 0.0f;
    float* cp = C + (size_t)mB * N + n;
#pragma unroll
    for (int j = 0; j < 8; ++j) {
      float v = acc[nt][j] + bs;
      if (act) v = tanhf(v);
      cp[(size_t)j * N] = v;
    }
  }
}

// ---------------------------------------------------------------------------
// Small helper kernels
// ---------------------------------------------------------------------------
__global__ void cvt_kernel(const float* __restrict__ src, bf16_t* __restrict__ dst, int n) {
  int i = blockIdx.x * blockDim.x + threadIdx.x;
  if (i < n) dst[i] = f2bf(src[i]);
}

// W_cat (4H x 3H) = [W_ih (4H x 2H) | W_hh (4H x H)], converted to bf16
__global__ void build_wcat_kernel(const float* __restrict__ W_ih, const float* __restrict__ W_hh,
                                  bf16_t* __restrict__ dst, int n) {
  int i = blockIdx.x * blockDim.x + threadIdx.x;
  if (i >= n) return;
  int row = i / (3 * Hq);
  int k   = i % (3 * Hq);
  float v = (k < 2 * Hq) ? W_ih[(size_t)row * (2 * Hq) + k]
                         : W_hh[(size_t)row * Hq + (k - 2 * Hq)];
  dst[i] = f2bf(v);
}

__global__ void add_kernel(const float* a, const float* b, float* d, int n) {
  int i = blockIdx.x * blockDim.x + threadIdx.x;
  if (i < n) d[i] = a[i] + b[i];
}

__global__ void init_state_kernel(const float* h0, const float* c0,
                                  float* h, float* c, float* prev, int n) {
  int i = blockIdx.x * blockDim.x + threadIdx.x;
  if (i < n) { h[i] = h0[i]; c[i] = c0[i]; prev[i] = 0.0f; }
}

// A_cat(b, 0:512)=emb[tgt[t,b]], (512:1024)=prev_out, (1024:1536)=h   (bf16)
__global__ void build_dec_in_kernel(const float* __restrict__ emb, const int* __restrict__ tgt,
                                    const float* __restrict__ prev, const float* __restrict__ h,
                                    bf16_t* __restrict__ acat, int t) {
  int i = blockIdx.x * blockDim.x + threadIdx.x;
  if (i >= Bq * 3 * Hq) return;
  int b = i / (3 * Hq);
  int k = i % (3 * Hq);
  float v;
  if (k < Hq) {
    int tok = tgt[t * Bq + b];
    v = emb[(size_t)tok * Hq + k];
  } else if (k < 2 * Hq) {
    v = prev[(size_t)b * Hq + (k - Hq)];
  } else {
    v = h[(size_t)b * Hq + (k - 2 * Hq)];
  }
  acat[i] = f2bf(v);
}

// gates layout: [i | f | g | o] each H wide
__global__ void lstm_kernel(const float* __restrict__ gates, float* h, float* c) {
  int i = blockIdx.x * blockDim.x + threadIdx.x;
  if (i >= Bq * Hq) return;
  int b = i / Hq, j = i % Hq;
  const float* g = gates + (size_t)b * 4 * Hq;
  float ig = sigf(g[j]);
  float fg = sigf(g[Hq + j]);
  float gg = tanhf(g[2 * Hq + j]);
  float og = sigf(g[3 * Hq + j]);
  float cn = fg * c[i] + ig * gg;
  c[i] = cn;
  h[i] = og * tanhf(cn);
}

// e[b,s] = dot(keys[b,s,:], h[b,:])  -- one wave32 per (b,s)
__global__ void attn_scores_kernel(const float* __restrict__ keys, const float* __restrict__ h,
                                   float* __restrict__ e) {
  int gid  = blockIdx.x * blockDim.x + threadIdx.x;
  int wave = gid >> 5, lane = gid & 31;
  if (wave >= Bq * Sq) return;
  int b = wave >> 7;            // S == 128
  const float* kp = keys + (size_t)wave * Hq;
  const float* hp = h + (size_t)b * Hq;
  float s = 0.0f;
  for (int i = lane; i < Hq; i += 32) s += kp[i] * hp[i];
#pragma unroll
  for (int off = 16; off > 0; off >>= 1) s += __shfl_down(s, off, 32);
  if (lane == 0) e[wave] = s;
}

// masked softmax over S=128, one 128-thread block per batch row
__global__ void softmax_kernel(const float* __restrict__ e, const int* __restrict__ src_len,
                               float* __restrict__ alpha) {
  __shared__ float red[Sq];
  int b = blockIdx.x, s = threadIdx.x;
  bool valid = s < src_len[b];
  float v = valid ? e[b * Sq + s] : -INFINITY;
  red[s] = v; __syncthreads();
  for (int off = Sq / 2; off > 0; off >>= 1) {
    if (s < off) red[s] = fmaxf(red[s], red[s + off]);
    __syncthreads();
  }
  float m = red[0]; __syncthreads();
  float ex = valid ? expf(v - m) : 0.0f;
  red[s] = ex; __syncthreads();
  for (int off = Sq / 2; off > 0; off >>= 1) {
    if (s < off) red[s] += red[s + off];
    __syncthreads();
  }
  alpha[b * Sq + s] = ex / red[0];
}

// a[b,e'] = sum_s alpha[b,s] * enc[b,s,e']  (coalesced across e')
__global__ void attn_ctx_kernel(const float* __restrict__ alpha, const float* __restrict__ enc,
                                float* __restrict__ a) {
  int i = blockIdx.x * blockDim.x + threadIdx.x;
  if (i >= Bq * EHq) return;
  int b = i / EHq, ee = i % EHq;
  const float* ap = alpha + b * Sq;
  const float* ep = enc + (size_t)b * Sq * EHq + ee;
  float s = 0.0f;
  for (int t = 0; t < Sq; ++t) s += ap[t] * ep[(size_t)t * EHq];
  a[i] = s;
}

// comb_in(b, 0:1024)=a, (1024:1536)=h   (bf16)
__global__ void build_comb_in_kernel(const float* __restrict__ a, const float* __restrict__ h,
                                     bf16_t* __restrict__ dst) {
  int i = blockIdx.x * blockDim.x + threadIdx.x;
  if (i >= Bq * (EHq + Hq)) return;
  int b = i / (EHq + Hq), k = i % (EHq + Hq);
  float v = (k < EHq) ? a[(size_t)b * EHq + k] : h[(size_t)b * Hq + (k - EHq)];
  dst[i] = f2bf(v);
}

__global__ void store_pred_kernel(const float* __restrict__ o, bf16_t* __restrict__ preds, int t) {
  int i = blockIdx.x * blockDim.x + threadIdx.x;
  if (i < Bq * Hq) preds[(size_t)t * Bq * Hq + i] = f2bf(o[i]);
}

// in-place log_softmax over V per row; one 256-thread block per row
__global__ void logsoftmax_kernel(float* __restrict__ out) {
  __shared__ float red[256];
  const int tid = threadIdx.x;
  float* p = out + (size_t)blockIdx.x * Vq;
  float m = -INFINITY;
  for (int i = tid; i < Vq; i += 256) m = fmaxf(m, p[i]);
  red[tid] = m; __syncthreads();
  for (int off = 128; off > 0; off >>= 1) {
    if (tid < off) red[tid] = fmaxf(red[tid], red[tid + off]);
    __syncthreads();
  }
  m = red[0]; __syncthreads();
  float s = 0.0f;
  for (int i = tid; i < Vq; i += 256) s += expf(p[i] - m);
  red[tid] = s; __syncthreads();
  for (int off = 128; off > 0; off >>= 1) {
    if (tid < off) red[tid] += red[tid + off];
    __syncthreads();
  }
  const float lse = m + logf(red[0]);
  for (int i = tid; i < Vq; i += 256) p[i] -= lse;
}

// ---------------------------------------------------------------------------
extern "C" void kernel_launch(void* const* d_in, const int* in_sizes, int n_in,
                              void* d_out, int out_size, void* d_ws, size_t ws_size,
                              hipStream_t stream) {
  const float* h0      = (const float*)d_in[0];
  const float* c0      = (const float*)d_in[1];
  const float* enc     = (const float*)d_in[2];
  const int*   src_len = (const int*)d_in[3];
  const int*   tgt     = (const int*)d_in[4];
  const float* emb     = (const float*)d_in[5];
  const float* W_ih    = (const float*)d_in[6];
  const float* W_hh    = (const float*)d_in[7];
  const float* b_ih    = (const float*)d_in[8];
  const float* b_hh    = (const float*)d_in[9];
  const float* W_att   = (const float*)d_in[10];
  const float* W_comb  = (const float*)d_in[11];
  const float* W_vocab = (const float*)d_in[12];
  float* out = (float*)d_out;

  const int H4 = 4 * Hq;     // 2048
  const int KC = 3 * Hq;     // 1536

  char* wp = (char*)d_ws;
  auto alloc = [&](size_t bytes) -> char* {
    char* p = wp; wp += (bytes + 255) & ~(size_t)255; return p;
  };

  bf16_t* Watt_bf  = (bf16_t*)alloc((size_t)Hq * EHq * 2);
  bf16_t* Wcat_bf  = (bf16_t*)alloc((size_t)H4 * KC * 2);
  bf16_t* Wcomb_bf = (bf16_t*)alloc((size_t)Hq * (EHq + Hq) * 2);
  bf16_t* Wvoc_bf  = (bf16_t*)alloc((size_t)Vq * Hq * 2);
  bf16_t* enc_bf   = (bf16_t*)alloc((size_t)Bq * Sq * EHq * 2);
  bf16_t* preds_bf = (bf16_t*)alloc((size_t)Tq * Bq * Hq * 2);
  bf16_t* acat_bf  = (bf16_t*)alloc((size_t)Bq * KC * 2);
  bf16_t* comb_bf  = (bf16_t*)alloc((size_t)Bq * (EHq + Hq) * 2);
  float* bias_s = (float*)alloc((size_t)H4 * 4);
  float* keys   = (float*)alloc((size_t)Bq * Sq * Hq * 4);
  float* hbuf   = (float*)alloc((size_t)Bq * Hq * 4);
  float* cbuf   = (float*)alloc((size_t)Bq * Hq * 4);
  float* pout   = (float*)alloc((size_t)Bq * Hq * 4);
  float* gates  = (float*)alloc((size_t)Bq * H4 * 4);
  float* evec   = (float*)alloc((size_t)Bq * Sq * 4);
  float* alpha  = (float*)alloc((size_t)Bq * Sq * 4);
  float* actx   = (float*)alloc((size_t)Bq * EHq * 4);

  const int TB = 256;
  auto g1 = [](int n) { return dim3((unsigned)((n + 255) / 256)); };

  // One-time (per call) weight conversions + state init
  cvt_kernel<<<g1(Hq * EHq), TB, 0, stream>>>(W_att, Watt_bf, Hq * EHq);
  build_wcat_kernel<<<g1(H4 * KC), TB, 0, stream>>>(W_ih, W_hh, Wcat_bf, H4 * KC);
  cvt_kernel<<<g1(Hq * (EHq + Hq)), TB, 0, stream>>>(W_comb, Wcomb_bf, Hq * (EHq + Hq));
  cvt_kernel<<<g1(Vq * Hq), TB, 0, stream>>>(W_vocab, Wvoc_bf, Vq * Hq);
  cvt_kernel<<<g1(Bq * Sq * EHq), TB, 0, stream>>>(enc, enc_bf, Bq * Sq * EHq);
  add_kernel<<<g1(H4), TB, 0, stream>>>(b_ih, b_hh, bias_s, H4);
  init_state_kernel<<<g1(Bq * Hq), TB, 0, stream>>>(h0, c0, hbuf, cbuf, pout, Bq * Hq);

  // keys = enc @ W_att^T : M=8192, N=512, K=1024  (LDS-staged path)
  wmma_gemm_lds_kernel<<<dim3(Hq / 128, (Bq * Sq) / 128), 256, 0, stream>>>(
      enc_bf, Watt_bf, nullptr, keys, Bq * Sq, Hq, EHq, 0);

  // Sequential decode: 48 steps
  for (int t = 0; t < Tq; ++t) {
    build_dec_in_kernel<<<g1(Bq * KC), TB, 0, stream>>>(emb, tgt, pout, hbuf, acat_bf, t);
    // gates = [x|prev|h] @ [W_ih|W_hh]^T + (b_ih + b_hh) : M=64, N=2048, K=1536
    wmma_gemm_kernel<<<dim3(H4 / 64, Bq / 64), 128, 0, stream>>>(
        acat_bf, Wcat_bf, bias_s, gates, Bq, H4, KC, 0);
    lstm_kernel<<<g1(Bq * Hq), TB, 0, stream>>>(gates, hbuf, cbuf);
    attn_scores_kernel<<<g1(Bq * Sq * 32), TB, 0, stream>>>(keys, hbuf, evec);
    softmax_kernel<<<dim3(Bq), dim3(Sq), 0, stream>>>(evec, src_len, alpha);
    attn_ctx_kernel<<<g1(Bq * EHq), TB, 0, stream>>>(alpha, enc, actx);
    build_comb_in_kernel<<<g1(Bq * (EHq + Hq)), TB, 0, stream>>>(actx, hbuf, comb_bf);
    // out = tanh([a|h] @ W_comb^T) : M=64, N=512, K=1536 ; writes prev_out
    wmma_gemm_kernel<<<dim3(Hq / 64, Bq / 64), 128, 0, stream>>>(
        comb_bf, Wcomb_bf, nullptr, pout, Bq, Hq, EHq + Hq, 1);
    store_pred_kernel<<<g1(Bq * Hq), TB, 0, stream>>>(pout, preds_bf, t);
  }

  // logits = preds @ W_vocab^T : M=3072, N=32000, K=512  (dominant, LDS path)
  wmma_gemm_lds_kernel<<<dim3(Vq / 128, (Tq * Bq) / 128), 256, 0, stream>>>(
      preds_bf, Wvoc_bf, nullptr, out, Tq * Bq, Vq, Hq, 0);

  // in-place log_softmax over V
  logsoftmax_kernel<<<dim3(Tq * Bq), TB, 0, stream>>>(out);
}
